// GraphEncoder_14955076125212
// MI455X (gfx1250) — matrix operand order repaired and checked by reading, not compiled, not run on hardware
//
#include <hip/hip_runtime.h>
#include <hip/hip_bf16.h>

typedef __attribute__((ext_vector_type(16))) _Float16 v16h;
typedef __attribute__((ext_vector_type(8)))  float    v8f;

// ---------------------------------------------------------------------------
// Degree / normalization
// ---------------------------------------------------------------------------
__global__ void deg_init_kernel(float* __restrict__ deg, long long n) {
    long long i = (long long)blockIdx.x * blockDim.x + threadIdx.x;
    if (i < n) deg[i] = 1.0f;   // self-loop contributes 1 to every degree
}

__global__ void deg_accum_kernel(const long long* __restrict__ dst, long long E,
                                 float* __restrict__ deg) {
    long long e = (long long)blockIdx.x * blockDim.x + threadIdx.x;
    if (e < E) {
        if (e + 4096 < E) __builtin_prefetch(&dst[e + 4096], 0, 0);
        atomicAdd(&deg[dst[e]], 1.0f);
    }
}

__global__ void deg_finalize_kernel(float* __restrict__ deg, long long n) {
    long long i = (long long)blockIdx.x * blockDim.x + threadIdx.x;
    if (i < n) {
        float v = deg[i];
        deg[i] = (v > 0.0f) ? rsqrtf(v) : 0.0f;  // deg -> dinv in place
    }
}

// ---------------------------------------------------------------------------
// GEMM1: H[N x 64] = X[N x 128] @ W1[128 x 64], output stored as f16.
// W1 is staged TRANSPOSED in LDS as wldsT[col][K] so each lane's 16-half
// B fragment is one contiguous 32B chunk -> 2x ds_load_b128 per fragment.
// Block = 256 threads = 8 waves; each wave owns one 16x16 output tile
// (2 row-tiles x 4 col-tiles per block).
// ---------------------------------------------------------------------------
__global__ void gemm1_wmma_kernel(const float* __restrict__ X,
                                  const float* __restrict__ W,
                                  _Float16* __restrict__ H, long long N) {
    __shared__ __align__(128) _Float16 wldsT[64 * 128];   // [col][K]
    for (int i = threadIdx.x; i < 128 * 64; i += blockDim.x) {
        int k = i >> 6, c = i & 63;           // W row-major [K=128][C=64]
        wldsT[c * 128 + k] = (_Float16)W[i];  // coalesced read, transposed write
    }
    __syncthreads();

    const int wave = threadIdx.x >> 5;
    const int lane = threadIdx.x & 31;
    const long long rowTile = (long long)blockIdx.x * 2 + (wave >> 2);
    const int colTile = wave & 3;
    if (rowTile * 16 >= N) return;            // wave-uniform: EXEC stays all-ones

    const long long R  = rowTile * 16;
    const int       Cb = colTile * 16;
    const long long mrow  = R + (lane & 15);
    const int       kbase = (lane < 16) ? 0 : 16;  // A-fragment K window base
    const int       g16   = (lane < 16) ? 0 : 16;  // B-fragment K group
    const int       ncol  = Cb + (lane & 15);
    const _Float16* wcol  = wldsT + ncol * 128;    // contiguous K for this lane

    v8f acc = {};
    #pragma unroll
    for (int k0 = 0; k0 < 128; k0 += 32) {
        v16h a;
        const float* xr = X + mrow * 128 + k0 + kbase;
        #pragma unroll
        for (int j = 0; j < 8; ++j) {         // elems 0..7: K=kbase+j, 8..15: K=kbase+8+j
            a[j]     = (_Float16)xr[j];
            a[j + 8] = (_Float16)xr[j + 8];
        }
        v16h b = *(const v16h*)(wcol + k0 + g16);   // 32B aligned, 2x ds_load_b128
        acc = __builtin_amdgcn_wmma_f32_16x16x32_f16(
            false, a, false, b, (short)0, acc, false, false);
    }

    const int rowAdd = (lane < 16) ? 0 : 8;
    #pragma unroll
    for (int r = 0; r < 8; ++r) {
        long long row = R + r + rowAdd;
        H[row * 64 + Cb + (lane & 15)] = (_Float16)acc[r];
    }
}

// ---------------------------------------------------------------------------
// Edge scatter, layer 1: agg[dst, :] += norm * H[src, :]   (64 f16 features)
// One wave per edge; each lane handles a half2 (coalesced 128B per edge).
// ---------------------------------------------------------------------------
__global__ void edge_agg1_kernel(const long long* __restrict__ src,
                                 const long long* __restrict__ dst,
                                 long long E,
                                 const _Float16* __restrict__ H,
                                 const float* __restrict__ dinv,
                                 float* __restrict__ agg) {
    const int lane = threadIdx.x & 31;
    const long long wid = (long long)blockIdx.x * (blockDim.x >> 5) + (threadIdx.x >> 5);
    if (wid >= E) return;
    if (wid + 2048 < E) {
        __builtin_prefetch(&src[wid + 2048], 0, 0);
        __builtin_prefetch(&dst[wid + 2048], 0, 0);
    }
    const long long s = src[wid];
    const long long d = dst[wid];
    const float norm = dinv[s] * dinv[d];

    const unsigned int* hrow = (const unsigned int*)(H + s * 64);
    union { unsigned int u; _Float16 h[2]; } cv;
    cv.u = hrow[lane];
    const float f0 = (float)cv.h[0];
    const float f1 = (float)cv.h[1];
    atomicAdd(&agg[d * 64 + 2 * lane],     norm * f0);
    atomicAdd(&agg[d * 64 + 2 * lane + 1], norm * f1);
}

// ---------------------------------------------------------------------------
// Fused self-loop + ReLU: agg[i,f] = relu(agg[i,f] + dinv[i]^2 * H[i,f])
// ---------------------------------------------------------------------------
__global__ void self_relu_kernel(float* __restrict__ agg,
                                 const _Float16* __restrict__ H,
                                 const float* __restrict__ dinv,
                                 long long total) {
    long long i = (long long)blockIdx.x * blockDim.x + threadIdx.x;
    if (i < total) {
        long long node = i >> 6;            // 64 features
        float di = dinv[node];
        float v = agg[i] + di * di * (float)H[i];
        agg[i] = (v > 0.0f) ? v : 0.0f;
    }
}

// ---------------------------------------------------------------------------
// GEMM2: Z[N x 32] = H1[N x 64] @ W2[64 x 32]  (f32 in, f16 compute, f32 out)
// W2 staged transposed in LDS: wldsT[col][K=64].
// Block = 8 waves: 4 row-tiles x 2 col-tiles.
// ---------------------------------------------------------------------------
__global__ void gemm2_wmma_kernel(const float* __restrict__ Hin,
                                  const float* __restrict__ W,
                                  float* __restrict__ Z, long long N) {
    __shared__ __align__(128) _Float16 wldsT[32 * 64];    // [col][K]
    for (int i = threadIdx.x; i < 64 * 32; i += blockDim.x) {
        int k = i >> 5, c = i & 31;           // W row-major [K=64][C=32]
        wldsT[c * 64 + k] = (_Float16)W[i];
    }
    __syncthreads();

    const int wave = threadIdx.x >> 5;
    const int lane = threadIdx.x & 31;
    const long long rowTile = (long long)blockIdx.x * 4 + (wave >> 1);
    const int colTile = wave & 1;
    if (rowTile * 16 >= N) return;            // wave-uniform

    const long long R  = rowTile * 16;
    const int       Cb = colTile * 16;
    const long long mrow  = R + (lane & 15);
    const int       kbase = (lane < 16) ? 0 : 16;
    const int       g16   = (lane < 16) ? 0 : 16;
    const int       ncol  = Cb + (lane & 15);
    const _Float16* wcol  = wldsT + ncol * 64;

    v8f acc = {};
    #pragma unroll
    for (int k0 = 0; k0 < 64; k0 += 32) {
        v16h a;
        const float* xr = Hin + mrow * 64 + k0 + kbase;
        #pragma unroll
        for (int j = 0; j < 8; ++j) {
            a[j]     = (_Float16)xr[j];
            a[j + 8] = (_Float16)xr[j + 8];
        }
        v16h b = *(const v16h*)(wcol + k0 + g16);
        acc = __builtin_amdgcn_wmma_f32_16x16x32_f16(
            false, a, false, b, (short)0, acc, false, false);
    }

    const int rowAdd = (lane < 16) ? 0 : 8;
    #pragma unroll
    for (int r = 0; r < 8; ++r) {
        long long row = R + r + rowAdd;
        Z[row * 32 + Cb + (lane & 15)] = acc[r];
    }
}

// ---------------------------------------------------------------------------
// Edge scatter, layer 2: out[dst, :] += norm * Z[src, :]   (32 f32 features)
// One wave per edge; one float per lane (coalesced 128B per edge).
// ---------------------------------------------------------------------------
__global__ void edge_agg2_kernel(const long long* __restrict__ src,
                                 const long long* __restrict__ dst,
                                 long long E,
                                 const float* __restrict__ Z,
                                 const float* __restrict__ dinv,
                                 float* __restrict__ out) {
    const int lane = threadIdx.x & 31;
    const long long wid = (long long)blockIdx.x * (blockDim.x >> 5) + (threadIdx.x >> 5);
    if (wid >= E) return;
    if (wid + 2048 < E) {
        __builtin_prefetch(&src[wid + 2048], 0, 0);
        __builtin_prefetch(&dst[wid + 2048], 0, 0);
    }
    const long long s = src[wid];
    const long long d = dst[wid];
    const float norm = dinv[s] * dinv[d];
    const float v = Z[s * 32 + lane];
    atomicAdd(&out[d * 32 + lane], norm * v);
}

// out[i,f] += dinv[i]^2 * Z[i,f]   (self-loop term, no atomics needed)
__global__ void self_add2_kernel(float* __restrict__ out,
                                 const float* __restrict__ Z,
                                 const float* __restrict__ dinv,
                                 long long total) {
    long long i = (long long)blockIdx.x * blockDim.x + threadIdx.x;
    if (i < total) {
        long long node = i >> 5;            // 32 features
        float di = dinv[node];
        out[i] += di * di * Z[i];
    }
}

// ---------------------------------------------------------------------------
// Host-side launch (graph-capture safe: kernels + hipMemsetAsync only)
// ---------------------------------------------------------------------------
extern "C" void kernel_launch(void* const* d_in, const int* in_sizes, int n_in,
                              void* d_out, int out_size, void* d_ws, size_t ws_size,
                              hipStream_t stream) {
    const float*     x   = (const float*)d_in[0];       // [N,128] f32
    const long long* ei  = (const long long*)d_in[1];   // [2,E]  i64
    const float*     W1  = (const float*)d_in[2];       // [128,64]
    const float*     W2  = (const float*)d_in[3];       // [64,32]
    float*           out = (float*)d_out;               // [N,32]

    const long long N = (long long)in_sizes[0] / 128;
    const long long E = (long long)in_sizes[1] / 2;
    const long long* srcI = ei;        // edge_index[0]
    const long long* dstI = ei + E;    // edge_index[1]

    // Workspace layout:
    //   dinv : N f32
    //   agg1 : N*64 f32   (layer-1 accumulator, becomes hidden1 after relu)
    //   h1   : N*64 f16   (X@W1, pre-aggregation; f16 halves gather traffic)
    //   z2   : N*32 f32   (hidden1@W2, pre-aggregation)
    float*     dinv = (float*)d_ws;
    float*     agg1 = dinv + N;
    _Float16*  h1   = (_Float16*)(agg1 + N * 64);
    float*     z2   = (float*)(h1 + N * 64);

    hipMemsetAsync(agg1, 0, sizeof(float) * (size_t)N * 64, stream);
    hipMemsetAsync(out,  0, sizeof(float) * (size_t)out_size, stream);

    const int T = 256;
    deg_init_kernel<<<(unsigned)((N + T - 1) / T), T, 0, stream>>>(dinv, N);
    deg_accum_kernel<<<(unsigned)((E + T - 1) / T), T, 0, stream>>>(dstI, E, dinv);
    deg_finalize_kernel<<<(unsigned)((N + T - 1) / T), T, 0, stream>>>(dinv, N);

    const long long rowTiles = (N + 15) / 16;           // 6250 for N=100000
    gemm1_wmma_kernel<<<(unsigned)((rowTiles + 1) / 2), 256, 0, stream>>>(x, W1, h1, N);

    const long long wavesPerBlock = 8;                  // 256 threads
    edge_agg1_kernel<<<(unsigned)((E + wavesPerBlock - 1) / wavesPerBlock), 256, 0, stream>>>(
        srcI, dstI, E, h1, dinv, agg1);

    self_relu_kernel<<<(unsigned)((N * 64 + T - 1) / T), T, 0, stream>>>(agg1, h1, dinv, N * 64);

    gemm2_wmma_kernel<<<(unsigned)((rowTiles + 3) / 4), 256, 0, stream>>>(agg1, W2, z2, N);

    edge_agg2_kernel<<<(unsigned)((E + wavesPerBlock - 1) / wavesPerBlock), 256, 0, stream>>>(
        srcI, dstI, E, z2, dinv, out);

    self_add2_kernel<<<(unsigned)((N * 32 + T - 1) / T), T, 0, stream>>>(out, z2, dinv, N * 32);
}
